// LlamaLayer_83442624627159
// MI455X (gfx1250) — matrix operand order repaired and checked by reading, not compiled, not run on hardware
//
#include <hip/hip_runtime.h>
#include <hip/hip_bf16.h>
#include <stdint.h>

#define DIM    2048
#define HEADS  16
#define HD     128
#define INTER  8192
#define BATCH  2
#define SEQ    2048
#define ROWS   (BATCH*SEQ)   /* 4096 token rows */
#define EPSF   1e-6f
#define NEGF   (-1e9f)

typedef __attribute__((ext_vector_type(16))) _Float16 v16h;
typedef __attribute__((ext_vector_type(8)))  _Float16 h8;
typedef __attribute__((ext_vector_type(8)))  float    v8f;

union V16U { v16h v; h8 h[2]; };

__device__ __forceinline__ v8f v8f_zero() {
    v8f z;
#pragma unroll
    for (int i = 0; i < 8; ++i) z[i] = 0.f;
    return z;
}

// ---------------------------------------------------------------------------
// CDNA5 async global->LDS copy (16B per lane), tracked by ASYNCcnt.
// LDS byte address = low 32 bits of the flat pointer (ISA 10.2: aperture
// mapping truncates flat LDS addresses to their low 32 bits).
// ---------------------------------------------------------------------------
__device__ __forceinline__ void async_ld_b128(uint32_t lds_addr, const void* gaddr) {
    asm volatile("global_load_async_to_lds_b128 %0, %1, off"
                 :: "v"(lds_addr), "v"((unsigned long long)(uintptr_t)gaddr)
                 : "memory");
}
__device__ __forceinline__ void wait_async0() {
    asm volatile("s_wait_asynccnt 0x0" ::: "memory");
}
__device__ __forceinline__ uint32_t lds_u32(const void* p) {
    return (uint32_t)(uintptr_t)p;
}

// ---------------------------------------------------------------------------
// fp32 -> f16 elementwise convert (weights, one pass; halves HBM bytes for
// every subsequent GEMM read on a bandwidth-bound layer)
// ---------------------------------------------------------------------------
__global__ void cvt_f32_f16(const float* __restrict__ s, _Float16* __restrict__ d, long n) {
    long i = (long)blockIdx.x * blockDim.x + threadIdx.x;
    const long stride = (long)gridDim.x * blockDim.x;
    for (; i < n; i += stride) d[i] = (_Float16)s[i];
}

// ---------------------------------------------------------------------------
// RMSNorm: one block per token row; fp32 math, f16 output (GEMM A operand)
// ---------------------------------------------------------------------------
__global__ __launch_bounds__(256) void rmsnorm_f16(const float* __restrict__ x,
                                                   const float* __restrict__ w,
                                                   _Float16* __restrict__ out) {
    __shared__ float red[8];
    const int row = blockIdx.x;
    const float* xr = x + (long)row * DIM;
    float ss = 0.f;
    for (int i = threadIdx.x; i < DIM; i += 256) { float v = xr[i]; ss += v * v; }
#pragma unroll
    for (int o = 16; o; o >>= 1) ss += __shfl_xor(ss, o, 32);
    if ((threadIdx.x & 31) == 0) red[threadIdx.x >> 5] = ss;
    __syncthreads();
    float tot = 0.f;
#pragma unroll
    for (int i = 0; i < 8; ++i) tot += red[i];
    const float r = rsqrtf(tot * (1.0f / DIM) + EPSF);
    _Float16* orow = out + (long)row * DIM;
    for (int i = threadIdx.x; i < DIM; i += 256) orow[i] = (_Float16)(xr[i] * r * w[i]);
}

// ---------------------------------------------------------------------------
// Double-buffered WMMA GEMM: C[M,N] = A[M,K](f16) x B[K,N](f16) (+res fp32)
// Block tile 128x128, K-step 32; 8 waves = 4(M) x 2(N); 8 WMMA/wave/step.
// A tile staged via global_load_async_to_lds_b128 (ASYNCcnt DMA, no VGPR
// round-trip); B tile loaded to VGPRs at loop top, transpose-scattered to LDS
// after compute; one barrier per step (double buffer). DMA + B global loads
// for tile t+1 overlap the WMMAs of tile t.
// ---------------------------------------------------------------------------
__global__ __launch_bounds__(256) void gemm_f16(const _Float16* __restrict__ A,
                                                const _Float16* __restrict__ Bm,
                                                const int M, const int N, const int K,
                                                const float* __restrict__ res,
                                                float* __restrict__ outF,
                                                _Float16* __restrict__ outH) {
    __shared__ __align__(16) _Float16 Al[2][128][40];  // [m][k]  (pad 8 halves)
    __shared__ __align__(16) _Float16 Bt[2][128][40];  // [n][k]  (transposed)
    const int tid  = threadIdx.x;
    const int lane = tid & 31, wave = tid >> 5;
    const int g = lane >> 4, l16 = lane & 15;
    const int wm = wave >> 1, wn = wave & 1;
    const int bm = blockIdx.y * 128, bn = blockIdx.x * 128;

    v8f acc[2][4];
#pragma unroll
    for (int mt = 0; mt < 2; ++mt)
#pragma unroll
        for (int nt = 0; nt < 4; ++nt) acc[mt][nt] = v8f_zero();

    const int arow = tid >> 1, acol = (tid & 1) * 16;   // 2 thr/row, 16 halves
    const int brow = tid >> 3, bcol = (tid & 7) * 16;   // 8 thr/row, 16 halves
    const _Float16* aSrc = A + (long)(bm + arow) * K + acol;
    const _Float16* bSrc = Bm + (long)brow * N + bn + bcol;
    const uint32_t aLds[2] = { lds_u32(&Al[0][arow][acol]), lds_u32(&Al[1][arow][acol]) };

    // ---- prologue: stage tile 0 into buffer 0
    async_ld_b128(aLds[0],      aSrc);
    async_ld_b128(aLds[0] + 16, aSrc + 8);
    {
        h8 b0 = *(const h8*)bSrc;
        h8 b1 = *(const h8*)(bSrc + 8);
#pragma unroll
        for (int i = 0; i < 8; ++i) Bt[0][bcol + i][brow]     = b0[i];
#pragma unroll
        for (int i = 0; i < 8; ++i) Bt[0][bcol + 8 + i][brow] = b1[i];
    }
    wait_async0();
    __syncthreads();

    const int nk = K / 32;
    for (int kt = 0; kt < nk; ++kt) {
        const int b = kt & 1;
        const bool more = (kt + 1) < nk;
        h8 bv0 = {}, bv1 = {};
        if (more) {  // issue tile t+1: DMA for A, global->VGPR for B
            const long ko = (long)(kt + 1) * 32;
            async_ld_b128(aLds[1 - b],      aSrc + ko);
            async_ld_b128(aLds[1 - b] + 16, aSrc + ko + 8);
            const _Float16* bs = bSrc + ko * N;
            bv0 = *(const h8*)bs;
            bv1 = *(const h8*)(bs + 8);
            __builtin_prefetch(aSrc + ko + 32, 0, 0);  // global_prefetch_b8: t+2
        }

        // ---- compute tile t from buffer b (overlaps the DMA above)
        V16U af[2];
#pragma unroll
        for (int mt = 0; mt < 2; ++mt) {
            const int r = wm * 32 + mt * 16 + l16;
            af[mt].h[0] = *(const h8*)&Al[b][r][8 * g];       // K = 8g..8g+7
            af[mt].h[1] = *(const h8*)&Al[b][r][16 + 8 * g];  // K = 16+8g..
        }
        V16U bf[4];
#pragma unroll
        for (int nt = 0; nt < 4; ++nt) {
            const int c = wn * 64 + nt * 16 + l16;
            bf[nt].h[0] = *(const h8*)&Bt[b][c][8 * g];       // K = 8g..8g+7
            bf[nt].h[1] = *(const h8*)&Bt[b][c][16 + 8 * g];  // K = 16+8g..
        }
#pragma unroll
        for (int mt = 0; mt < 2; ++mt)
#pragma unroll
            for (int nt = 0; nt < 4; ++nt)
                acc[mt][nt] = __builtin_amdgcn_wmma_f32_16x16x32_f16(
                    false, af[mt].v, false, bf[nt].v, (short)0, acc[mt][nt], false, false);

        if (more) {  // transpose-scatter B tile t+1 (global latency now hidden)
#pragma unroll
            for (int i = 0; i < 8; ++i) Bt[1 - b][bcol + i][brow]     = bv0[i];
#pragma unroll
            for (int i = 0; i < 8; ++i) Bt[1 - b][bcol + 8 + i][brow] = bv1[i];
        }
        wait_async0();      // this wave's DMA contribution to buffer 1-b landed
        __syncthreads();    // single barrier per step (double buffered)
    }

    // ---- epilogue: C-layout row = 8g+j, col = lane&15
#pragma unroll
    for (int mt = 0; mt < 2; ++mt)
#pragma unroll
        for (int nt = 0; nt < 4; ++nt) {
            const int c = bn + wn * 64 + nt * 16 + l16;
#pragma unroll
            for (int j = 0; j < 8; ++j) {
                const int r = bm + wm * 32 + mt * 16 + 8 * g + j;
                float v = acc[mt][nt][j];
                if (res)  v += res[(long)r * N + c];
                if (outF) outF[(long)r * N + c] = v;
                if (outH) outH[(long)r * N + c] = (_Float16)v;
            }
        }
}

// ---------------------------------------------------------------------------
// Flash attention: grid (S/128, HEADS, B); block 256 = 8 waves.
// Each wave owns 16 q rows x full head_dim(128): 8 f32 accumulators.
// KV tiles of 32; online softmax; never materializes S x S scores.
// Q/K staged via async DMA to LDS; V transposed manually (DMA can't scatter).
// ---------------------------------------------------------------------------
__global__ __launch_bounds__(256) void flash_attn(const _Float16* __restrict__ Q,
                                                  const _Float16* __restrict__ K,
                                                  const _Float16* __restrict__ V,
                                                  _Float16* __restrict__ O) {
    __shared__ __align__(16) _Float16 Ql[128][136];  // [q][hd]
    __shared__ __align__(16) _Float16 Kl[32][136];   // [kv][hd] (B-frag does T)
    __shared__ __align__(16) _Float16 Vl[128][40];   // [hd][kv] (transposed)
    __shared__ __align__(16) _Float16 Pl[8][16][40]; // per-wave P exchange

    const int tid  = threadIdx.x;
    const int lane = tid & 31, wave = tid >> 5;
    const int g = lane >> 4, l16 = lane & 15;
    const int q0 = blockIdx.x * 128;
    const long base = ((long)blockIdx.z * SEQ) * DIM + (long)blockIdx.y * HD;

    {   // stage Q block (128 x 128 halves) via async DMA
        const int r = tid >> 1, c0 = (tid & 1) * 64;
        const _Float16* src = Q + base + (long)(q0 + r) * DIM + c0;
        const uint32_t ql = lds_u32(&Ql[r][c0]);
#pragma unroll
        for (int i = 0; i < 8; ++i) async_ld_b128(ql + 16 * i, src + 8 * i);
    }
    wait_async0();
    __syncthreads();

    V16U qf[4];
    {   // A-fragments for this wave's 16 q rows (4 K-steps of 32 over head_dim)
        const int r = wave * 16 + l16;
#pragma unroll
        for (int f = 0; f < 4; ++f) {
            qf[f].h[0] = *(const h8*)&Ql[r][32 * f + 8 * g];
            qf[f].h[1] = *(const h8*)&Ql[r][32 * f + 16 + 8 * g];
        }
    }

    v8f oacc[8];
#pragma unroll
    for (int t = 0; t < 8; ++t) oacc[t] = v8f_zero();
    float mrow[8], lrow[8];
#pragma unroll
    for (int j = 0; j < 8; ++j) { mrow[j] = -__builtin_inff(); lrow[j] = 0.f; }

    const float rs = 0.088388347648318447f;  // 1/sqrt(128)
    const int kv_end = q0 + 128;             // causal block bound

    for (int kv0 = 0; kv0 < kv_end; kv0 += 32) {
        {   // stage K via async DMA, V via transpose scatter
            const int kr = tid >> 3, kc = (tid & 7) * 16;
            const _Float16* ks = K + base + (long)(kv0 + kr) * DIM + kc;
            const uint32_t kl = lds_u32(&Kl[kr][kc]);
            async_ld_b128(kl,      ks);
            async_ld_b128(kl + 16, ks + 8);
            const _Float16* vs = V + base + (long)(kv0 + kr) * DIM + kc;
            h8 v0 = *(const h8*)vs;
            h8 v1 = *(const h8*)(vs + 8);
#pragma unroll
            for (int i = 0; i < 8; ++i) Vl[kc + i][kr]     = v0[i];
#pragma unroll
            for (int i = 0; i < 8; ++i) Vl[kc + 8 + i][kr] = v1[i];
            __builtin_prefetch(ks + 32 * DIM, 0, 0);  // next KV tile hint
        }
        wait_async0();
        __syncthreads();

        // scores S = Q K^T (2 n-tiles x 4 k-steps)
        v8f s[2];
        s[0] = v8f_zero(); s[1] = v8f_zero();
#pragma unroll
        for (int nt = 0; nt < 2; ++nt) {
            const int c = nt * 16 + l16;  // kv column owned by this lane
#pragma unroll
            for (int f = 0; f < 4; ++f) {
                V16U bf;  // B[kdim][kv] = K[kv][kdim] -> row read of Kl = free T
                bf.h[0] = *(const h8*)&Kl[c][32 * f + 8 * g];
                bf.h[1] = *(const h8*)&Kl[c][32 * f + 16 + 8 * g];
                s[nt] = __builtin_amdgcn_wmma_f32_16x16x32_f16(
                    false, qf[f].v, false, bf.v, (short)0, s[nt], false, false);
            }
        }

        // scale + causal additive mask
#pragma unroll
        for (int nt = 0; nt < 2; ++nt) {
            const int cg = kv0 + nt * 16 + l16;
#pragma unroll
            for (int j = 0; j < 8; ++j) {
                const int rg = q0 + wave * 16 + 8 * g + j;
                float sv = s[nt][j] * rs;
                if (cg > rg) sv += NEGF;
                s[nt][j] = sv;
            }
        }

        // online softmax per row (rows distributed as 8g+j across 16-lane halves)
#pragma unroll
        for (int j = 0; j < 8; ++j) {
            float mx = fmaxf(s[0][j], s[1][j]);
#pragma unroll
            for (int o = 8; o; o >>= 1) mx = fmaxf(mx, __shfl_xor(mx, o, 16));
            const float mnew = fmaxf(mrow[j], mx);
            const float corr = __expf(mrow[j] - mnew);
            const float p0 = __expf(s[0][j] - mnew);
            const float p1 = __expf(s[1][j] - mnew);
            float ps = p0 + p1;
#pragma unroll
            for (int o = 8; o; o >>= 1) ps += __shfl_xor(ps, o, 16);
            lrow[j] = lrow[j] * corr + ps;
            mrow[j] = mnew;
#pragma unroll
            for (int t = 0; t < 8; ++t) oacc[t][j] *= corr;
            Pl[wave][8 * g + j][l16]      = (_Float16)p0;  // C-layout -> LDS
            Pl[wave][8 * g + j][16 + l16] = (_Float16)p1;
        }
        // wave-private LDS exchange: only this wave's DS ops matter
        asm volatile("s_wait_dscnt 0x0" ::: "memory");

        V16U pf;  // re-read P in A-fragment layout (16 q rows x 32 kv)
        pf.h[0] = *(const h8*)&Pl[wave][l16][8 * g];
        pf.h[1] = *(const h8*)&Pl[wave][l16][16 + 8 * g];
#pragma unroll
        for (int t = 0; t < 8; ++t) {
            const int c = t * 16 + l16;  // head-dim column
            V16U bf;
            bf.h[0] = *(const h8*)&Vl[c][8 * g];
            bf.h[1] = *(const h8*)&Vl[c][16 + 8 * g];
            oacc[t] = __builtin_amdgcn_wmma_f32_16x16x32_f16(
                false, pf.v, false, bf.v, (short)0, oacc[t], false, false);
        }
        __syncthreads();
    }

    // normalize and write f16 (feeds O-projection GEMM)
#pragma unroll
    for (int t = 0; t < 8; ++t)
#pragma unroll
        for (int j = 0; j < 8; ++j) {
            const int rg = q0 + wave * 16 + 8 * g + j;
            const float ov = oacc[t][j] / lrow[j];
            O[base + (long)rg * DIM + t * 16 + l16] = (_Float16)ov;
        }
}

// ---------------------------------------------------------------------------
// SwiGLU elementwise: t = silu(g) * u  (f16 in/out; fp32 math)
// ---------------------------------------------------------------------------
__global__ void swiglu(const _Float16* __restrict__ gt, const _Float16* __restrict__ ut,
                       _Float16* __restrict__ t, long n) {
    long i = (long)blockIdx.x * blockDim.x + threadIdx.x;
    const long stride = (long)gridDim.x * blockDim.x;
    for (; i < n; i += stride) {
        const float gv = (float)gt[i];
        const float uv = (float)ut[i];
        t[i] = (_Float16)((gv / (1.f + __expf(-gv))) * uv);
    }
}

// ---------------------------------------------------------------------------
extern "C" void kernel_launch(void* const* d_in, const int* in_sizes, int n_in,
                              void* d_out, int out_size, void* d_ws, size_t ws_size,
                              hipStream_t stream) {
    (void)in_sizes; (void)n_in; (void)out_size; (void)ws_size;
    const float* x   = (const float*)d_in[0];
    // d_in[1] = mask: causal, recomputed analytically in-kernel
    const float* wan = (const float*)d_in[2];
    const float* wq  = (const float*)d_in[3];
    const float* wk  = (const float*)d_in[4];
    const float* wv  = (const float*)d_in[5];
    const float* wo  = (const float*)d_in[6];
    const float* wfn = (const float*)d_in[7];
    const float* wg  = (const float*)d_in[8];
    const float* wu  = (const float*)d_in[9];
    const float* wd  = (const float*)d_in[10];
    float* out = (float*)d_out;

    char* cur = (char*)d_ws;
    auto allocH = [&](size_t n) -> _Float16* {
        _Float16* p = (_Float16*)cur;
        cur += ((n * sizeof(_Float16) + 255) & ~(size_t)255);
        return p;
    };
    auto allocF = [&](size_t n) -> float* {
        float* p = (float*)cur;
        cur += ((n * sizeof(float) + 255) & ~(size_t)255);
        return p;
    };

    _Float16* wq16 = allocH((size_t)DIM * DIM);
    _Float16* wk16 = allocH((size_t)DIM * DIM);
    _Float16* wv16 = allocH((size_t)DIM * DIM);
    _Float16* wo16 = allocH((size_t)DIM * DIM);
    _Float16* wg16 = allocH((size_t)DIM * INTER);
    _Float16* wu16 = allocH((size_t)DIM * INTER);
    _Float16* wd16 = allocH((size_t)INTER * DIM);
    _Float16* xn16 = allocH((size_t)ROWS * DIM);
    _Float16* q16  = allocH((size_t)ROWS * DIM);
    _Float16* k16  = allocH((size_t)ROWS * DIM);
    _Float16* v16  = allocH((size_t)ROWS * DIM);
    float*    h32  = allocF((size_t)ROWS * DIM);
    _Float16* g16  = allocH((size_t)ROWS * INTER);
    _Float16* u16  = allocH((size_t)ROWS * INTER);
    // stream-ordered buffer reuse:
    _Float16* ao16 = xn16;  // attn-out: xn consumed by QKV GEMMs before this
    _Float16* hn16 = k16;   // ffn-norm: k consumed by attention before this
    _Float16* t16  = g16;   // swiglu out overwrites gate in-place (elementwise)

    const dim3 blk(256);
    // weight conversions (one pass each)
    cvt_f32_f16<<<2048, 256, 0, stream>>>(wq, wq16, (long)DIM * DIM);
    cvt_f32_f16<<<2048, 256, 0, stream>>>(wk, wk16, (long)DIM * DIM);
    cvt_f32_f16<<<2048, 256, 0, stream>>>(wv, wv16, (long)DIM * DIM);
    cvt_f32_f16<<<2048, 256, 0, stream>>>(wo, wo16, (long)DIM * DIM);
    cvt_f32_f16<<<4096, 256, 0, stream>>>(wg, wg16, (long)DIM * INTER);
    cvt_f32_f16<<<4096, 256, 0, stream>>>(wu, wu16, (long)DIM * INTER);
    cvt_f32_f16<<<4096, 256, 0, stream>>>(wd, wd16, (long)INTER * DIM);

    // attn norm -> QKV -> flash attention -> O-proj (+residual x)
    rmsnorm_f16<<<ROWS, 256, 0, stream>>>(x, wan, xn16);
    const dim3 gD(DIM / 128, ROWS / 128);
    gemm_f16<<<gD, blk, 0, stream>>>(xn16, wq16, ROWS, DIM, DIM, nullptr, nullptr, q16);
    gemm_f16<<<gD, blk, 0, stream>>>(xn16, wk16, ROWS, DIM, DIM, nullptr, nullptr, k16);
    gemm_f16<<<gD, blk, 0, stream>>>(xn16, wv16, ROWS, DIM, DIM, nullptr, nullptr, v16);
    const dim3 gA(SEQ / 128, HEADS, BATCH);
    flash_attn<<<gA, blk, 0, stream>>>(q16, k16, v16, ao16);
    gemm_f16<<<gD, blk, 0, stream>>>(ao16, wo16, ROWS, DIM, DIM, x, h32, nullptr);

    // ffn norm -> gate/up -> swiglu -> down (+residual h) -> fp32 out
    rmsnorm_f16<<<ROWS, 256, 0, stream>>>(h32, wfn, hn16);
    const dim3 gI(INTER / 128, ROWS / 128);
    gemm_f16<<<gI, blk, 0, stream>>>(hn16, wg16, ROWS, INTER, DIM, nullptr, nullptr, g16);
    gemm_f16<<<gI, blk, 0, stream>>>(hn16, wu16, ROWS, INTER, DIM, nullptr, nullptr, u16);
    swiglu<<<4096, 256, 0, stream>>>(g16, u16, t16, (long)ROWS * INTER);
    gemm_f16<<<gD, blk, 0, stream>>>(t16, wd16, ROWS, DIM, INTER, h32, out, nullptr);
}
